// MultiHeadAttention_18665927868945
// MI455X (gfx1250) — matrix operand order repaired and checked by reading, not compiled, not run on hardware
//
#include <hip/hip_runtime.h>
#include <hip/hip_bf16.h>
#include <stdint.h>

// ---------------------------------------------------------------------------
// MHA forward for MI455X (gfx1250), wave32 + WMMA f32_16x16x32_f16.
// B=4, S=2048, D_MODEL=1024, H=16, Dk=Dv=64.
// Outputs: [out (B,S,1024) f32 | attn (B,H,S,S) f32] concatenated in d_out.
// GEMMs are double-buffered: global->reg loads for tile t+1 issue before the
// WMMA phase of tile t, so HBM latency is hidden behind matrix math.
// ---------------------------------------------------------------------------

typedef _Float16 v16h __attribute__((ext_vector_type(16)));
typedef _Float16 v8h  __attribute__((ext_vector_type(8)));
typedef float    v8f  __attribute__((ext_vector_type(8)));
typedef float    v4f  __attribute__((ext_vector_type(4)));

#define SEQ   2048
#define DMODEL 1024
#define NHEAD 16
#define HDIM  64

// ------------------------------ f32 -> f16 convert -------------------------
__global__ __launch_bounds__(256) void cvt_f32_to_f16_kernel(
    const float* __restrict__ in, _Float16* __restrict__ out, int n) {
  int i = blockIdx.x * 256 + threadIdx.x;
  if (i < n) out[i] = (_Float16)in[i];
}

// ------------------------------ WMMA tiled GEMM ----------------------------
// C(MxN) = A(MxK) * B(KxN), f32 accumulation via v_wmma_f32_16x16x32_f16.
// AMODE: 0 = A is f16, 1 = A is f32 (converted while staging into LDS)
// BMODE: 0 = B stored [k][n] row-major (transpose-stage into LDS[n][k])
//        1 = B stored [n][k] row-major (direct copy; used for K^T in scores)
// OMODE: 0 = f32 plain [m*ldc+n]
//        1 = f16 head-split:  out[((b*H+h)*S+s)*64+d], m=b*S+s, n=h*64+d
//        2 = f32 scores: v*scale, mask -> -1e9, out per-z [q*ldc+k]
//        3 = f16 context:  out[(b*S+row)*1024 + h*64 + col], z=b*H+h
template<int BM, int BN, int BK, int WROWS, int WCOLS, int AMODE, int BMODE, int OMODE>
__global__ __launch_bounds__(256) void gemm_wmma_kernel(
    const void* __restrict__ Aab, const _Float16* __restrict__ Bb,
    void* __restrict__ Ob, const unsigned char* __restrict__ mask,
    int Kd, int lda, int ldb, int ldc,
    long long strideAz, long long strideBz, long long strideOz, float scale) {
  constexpr int RM  = BM / (WROWS * 16);
  constexpr int RN  = BN / (WCOLS * 16);
  constexpr int PAD = 8;                      // 16B pad: keeps v8h alignment
  constexpr int LDA_S = BK + PAD;
  constexpr int ACH = (BM * BK / 8) / 256;    // 16B chunks per thread for A
  constexpr int BCH = ((BMODE == 0 ? BK * BN : BN * BK) / 8) / 256;

  __shared__ __attribute__((aligned(16))) _Float16 As[BM * LDA_S];
  __shared__ __attribute__((aligned(16))) _Float16 Bs[BN * LDA_S];

  const int tid   = threadIdx.x;
  const int lane  = tid & 31;
  const int wave  = tid >> 5;
  const int waveM = wave / WCOLS;
  const int waveN = wave % WCOLS;
  const int z     = blockIdx.z;
  const int tileM = blockIdx.x * BM;
  const int tileN = blockIdx.y * BN;

  const size_t zA = (size_t)z * (size_t)strideAz;
  const _Float16* __restrict__ Bg = Bb + (size_t)z * (size_t)strideBz;

  v8f acc[RM][RN];
#pragma unroll
  for (int i = 0; i < RM; ++i)
#pragma unroll
    for (int j = 0; j < RN; ++j) acc[i][j] = (v8f)0.0f;

  // register staging buffers for the next K-tile (double buffer)
  v8h aReg[ACH];
  v4f aRegF[AMODE == 1 ? 2 * ACH : 1];
  v8h bReg[BCH];

  auto loadTile = [&](int kt) {
    if constexpr (AMODE == 0) {
      const _Float16* __restrict__ Ag = (const _Float16*)Aab + zA;
#pragma unroll
      for (int u = 0; u < ACH; ++u) {
        int c = tid + u * 256;
        int r = c / (BK / 8), kc = (c % (BK / 8)) * 8;
        aReg[u] = *(const v8h*)(Ag + (size_t)(tileM + r) * lda + kt + kc);
      }
    } else {
      const float* __restrict__ Ag = (const float*)Aab + zA;
#pragma unroll
      for (int u = 0; u < ACH; ++u) {
        int c = tid + u * 256;
        int r = c / (BK / 8), kc = (c % (BK / 8)) * 8;
        const v4f* p = (const v4f*)(Ag + (size_t)(tileM + r) * lda + kt + kc);
        aRegF[2 * u] = p[0]; aRegF[2 * u + 1] = p[1];
      }
    }
    if constexpr (BMODE == 0) {   // global [k][n]
#pragma unroll
      for (int u = 0; u < BCH; ++u) {
        int c = tid + u * 256;
        int kr = c / (BN / 8), n0 = (c % (BN / 8)) * 8;
        bReg[u] = *(const v8h*)(Bg + (size_t)(kt + kr) * ldb + tileN + n0);
      }
    } else {                      // global [n][k]
#pragma unroll
      for (int u = 0; u < BCH; ++u) {
        int c = tid + u * 256;
        int r = c / (BK / 8), kc = (c % (BK / 8)) * 8;
        bReg[u] = *(const v8h*)(Bg + (size_t)(tileN + r) * ldb + kt + kc);
      }
    }
  };

  auto storeTile = [&]() {
    if constexpr (AMODE == 0) {
#pragma unroll
      for (int u = 0; u < ACH; ++u) {
        int c = tid + u * 256;
        int r = c / (BK / 8), kc = (c % (BK / 8)) * 8;
        *(v8h*)&As[r * LDA_S + kc] = aReg[u];
      }
    } else {
#pragma unroll
      for (int u = 0; u < ACH; ++u) {
        int c = tid + u * 256;
        int r = c / (BK / 8), kc = (c % (BK / 8)) * 8;
        v8h h;
#pragma unroll
        for (int t = 0; t < 4; ++t) {
          h[t]     = (_Float16)aRegF[2 * u][t];
          h[4 + t] = (_Float16)aRegF[2 * u + 1][t];
        }
        *(v8h*)&As[r * LDA_S + kc] = h;
      }
    }
    if constexpr (BMODE == 0) {   // transpose into LDS [n][k]
#pragma unroll
      for (int u = 0; u < BCH; ++u) {
        int c = tid + u * 256;
        int kr = c / (BN / 8), n0 = (c % (BN / 8)) * 8;
#pragma unroll
        for (int t = 0; t < 8; ++t) Bs[(n0 + t) * LDA_S + kr] = bReg[u][t];
      }
    } else {
#pragma unroll
      for (int u = 0; u < BCH; ++u) {
        int c = tid + u * 256;
        int r = c / (BK / 8), kc = (c % (BK / 8)) * 8;
        *(v8h*)&Bs[r * LDA_S + kc] = bReg[u];
      }
    }
  };

  loadTile(0);
  for (int kt = 0; kt < Kd; kt += BK) {
    storeTile();                 // s_wait_loadcnt lands here, after full tile
    __syncthreads();
    if (kt + BK < Kd) loadTile(kt + BK);   // overlap with WMMA phase below
    if (kt + 2 * BK < Kd) {      // L2-side prefetch of tile t+2
      if constexpr (AMODE == 0)
        __builtin_prefetch((const _Float16*)Aab + zA +
                           (size_t)(tileM + (tid & (BM - 1))) * lda + kt + 2 * BK, 0, 1);
      else
        __builtin_prefetch((const float*)Aab + zA +
                           (size_t)(tileM + (tid & (BM - 1))) * lda + kt + 2 * BK, 0, 1);
      __builtin_prefetch(Bg + (size_t)(kt + 2 * BK + (tid & (BK - 1))) * ldb + tileN, 0, 1);
    }

    // ---- compute: per wave RM x RN 16x16 tiles, k in steps of 32 ----
#pragma unroll
    for (int kk = 0; kk < BK / 32; ++kk) {
      const int lhi = (lane < 16) ? 0 : 1;
      v16h afrag[RM], bfrag[RN];
#pragma unroll
      for (int i = 0; i < RM; ++i) {
        int m  = waveM * RM * 16 + i * 16 + (lane & 15);
        int kb = kk * 32 + lhi * 8;     // lanes 0-15: K 0..7/16..23; 16-31: 8..15/24..31
        v8h lo = *(const v8h*)&As[m * LDA_S + kb];
        v8h hi = *(const v8h*)&As[m * LDA_S + kb + 16];
        afrag[i] = __builtin_shufflevector(lo, hi, 0,1,2,3,4,5,6,7,8,9,10,11,12,13,14,15);
      }
#pragma unroll
      for (int j = 0; j < RN; ++j) {
        int n  = waveN * RN * 16 + j * 16 + (lane & 15);
        int kb = kk * 32 + lhi * 8;     // lanes 0-15: K 0..7/16..23; 16-31: 8..15/24..31
        v8h lo = *(const v8h*)&Bs[n * LDA_S + kb];
        v8h hi = *(const v8h*)&Bs[n * LDA_S + kb + 16];
        bfrag[j] = __builtin_shufflevector(lo, hi, 0,1,2,3,4,5,6,7,8,9,10,11,12,13,14,15);
      }
#pragma unroll
      for (int i = 0; i < RM; ++i)
#pragma unroll
        for (int j = 0; j < RN; ++j)
          acc[i][j] = __builtin_amdgcn_wmma_f32_16x16x32_f16(
              false, afrag[i], false, bfrag[j], (short)0, acc[i][j], false, false);
    }
    __syncthreads();
  }

  // ---- epilogue: C/D layout — VGPR r: lanes 0-15 => M=r, lanes 16-31 => M=r+8
#pragma unroll
  for (int i = 0; i < RM; ++i)
#pragma unroll
    for (int j = 0; j < RN; ++j)
#pragma unroll
      for (int r = 0; r < 8; ++r) {
        int row = tileM + waveM * RM * 16 + i * 16 + ((lane < 16) ? r : r + 8);
        int col = tileN + waveN * RN * 16 + j * 16 + (lane & 15);
        float v = acc[i][j][r];
        if constexpr (OMODE == 0) {
          ((float*)Ob)[(size_t)row * ldc + col] = v;
        } else if constexpr (OMODE == 1) {
          int b = row >> 11, s = row & (SEQ - 1);
          int h = col >> 6,  d = col & (HDIM - 1);
          ((_Float16*)Ob)[(((size_t)b * NHEAD + h) * SEQ + s) * HDIM + d] = (_Float16)v;
        } else if constexpr (OMODE == 2) {
          v *= scale;
          const unsigned char* mz = mask + (size_t)(z >> 4) * SEQ * SEQ;
          if (mz[(size_t)row * SEQ + col]) v = -1e9f;
          float* O = (float*)Ob + (size_t)z * (size_t)strideOz;
          O[(size_t)row * ldc + col] = v;
        } else {  // OMODE == 3
          int b = z >> 4, h = z & (NHEAD - 1);
          ((_Float16*)Ob)[((size_t)b * SEQ + row) * DMODEL + (size_t)h * HDIM + col] =
              (_Float16)v;
        }
      }
}

// ------------------------------ row softmax (len 2048) ---------------------
__global__ __launch_bounds__(256) void softmax_rows_kernel(float* __restrict__ attn) {
  __shared__ float sdata[256];
  const int tid = threadIdx.x;
  float* row = attn + (size_t)blockIdx.x * SEQ;
  float v[8];
  float m = -3.4e38f;
#pragma unroll
  for (int k = 0; k < 8; ++k) { v[k] = row[tid + k * 256]; m = fmaxf(m, v[k]); }
  sdata[tid] = m; __syncthreads();
  for (int s = 128; s > 0; s >>= 1) {
    if (tid < s) sdata[tid] = fmaxf(sdata[tid], sdata[tid + s]);
    __syncthreads();
  }
  m = sdata[0]; __syncthreads();
  float sum = 0.0f;
#pragma unroll
  for (int k = 0; k < 8; ++k) { v[k] = __expf(v[k] - m); sum += v[k]; }
  sdata[tid] = sum; __syncthreads();
  for (int s = 128; s > 0; s >>= 1) {
    if (tid < s) sdata[tid] += sdata[tid + s];
    __syncthreads();
  }
  float inv = 1.0f / sdata[0];
#pragma unroll
  for (int k = 0; k < 8; ++k) row[tid + k * 256] = v[k] * inv;
}

// --------------------- residual add + LayerNorm (row=1024) -----------------
__global__ __launch_bounds__(256) void residual_ln_kernel(
    const float* __restrict__ proj, const float* __restrict__ resid,
    float* __restrict__ out) {
  __shared__ float s1[256];
  __shared__ float s2[256];
  const int tid = threadIdx.x;
  const size_t base = (size_t)blockIdx.x * DMODEL;
  float v[4]; float sum = 0.0f, ssq = 0.0f;
#pragma unroll
  for (int k = 0; k < 4; ++k) {
    v[k] = proj[base + tid + k * 256] + resid[base + tid + k * 256];
    sum += v[k]; ssq += v[k] * v[k];
  }
  s1[tid] = sum; s2[tid] = ssq; __syncthreads();
  for (int s = 128; s > 0; s >>= 1) {
    if (tid < s) { s1[tid] += s1[tid + s]; s2[tid] += s2[tid + s]; }
    __syncthreads();
  }
  float mu  = s1[0] * (1.0f / DMODEL);
  float var = s2[0] * (1.0f / DMODEL) - mu * mu;
  float rstd = rsqrtf(var + 1e-5f);
#pragma unroll
  for (int k = 0; k < 4; ++k)
    out[base + tid + k * 256] = (v[k] - mu) * rstd;
}

// ------------------------------ launcher -----------------------------------
extern "C" void kernel_launch(void* const* d_in, const int* in_sizes, int n_in,
                              void* d_out, int out_size, void* d_ws, size_t ws_size,
                              hipStream_t stream) {
  (void)in_sizes; (void)n_in; (void)out_size; (void)ws_size;
  const float* inQ = (const float*)d_in[0];
  const float* inK = (const float*)d_in[1];
  const float* inV = (const float*)d_in[2];
  const unsigned char* mask = (const unsigned char*)d_in[3];  // all-false bool
  const float* WQ = (const float*)d_in[4];
  const float* WK = (const float*)d_in[5];
  const float* WV = (const float*)d_in[6];
  const float* WO = (const float*)d_in[7];

  const int M   = 4 * SEQ;            // 8192 rows
  const int NIN = M * DMODEL;         // 8,388,608
  const int NW  = DMODEL * DMODEL;    // 1,048,576

  char* w = (char*)d_ws;
  _Float16* inq16 = (_Float16*)(w + 0);
  _Float16* ink16 = (_Float16*)(w + 16777216);
  _Float16* inv16 = (_Float16*)(w + 33554432);
  _Float16* wq16  = (_Float16*)(w + 50331648);
  _Float16* wk16  = (_Float16*)(w + 52428800);
  _Float16* wv16  = (_Float16*)(w + 54525952);
  _Float16* wo16  = (_Float16*)(w + 56623104);
  _Float16* Qh    = (_Float16*)(w + 58720256);   // [B,H,S,64] f16
  _Float16* Kh    = (_Float16*)(w + 75497472);
  _Float16* Vh    = (_Float16*)(w + 92274688);
  _Float16* ctx16 = inq16;                        // reuse after Q projection
  float*    proj  = (float*)ink16;                // reuse 32MB after K/V proj

  float* out  = (float*)d_out;
  float* attn = out + (size_t)M * DMODEL;         // +8,388,608 floats

  // 1) convert inputs + weights to f16
  cvt_f32_to_f16_kernel<<<(NIN + 255) / 256, 256, 0, stream>>>(inQ, inq16, NIN);
  cvt_f32_to_f16_kernel<<<(NIN + 255) / 256, 256, 0, stream>>>(inK, ink16, NIN);
  cvt_f32_to_f16_kernel<<<(NIN + 255) / 256, 256, 0, stream>>>(inV, inv16, NIN);
  cvt_f32_to_f16_kernel<<<(NW + 255) / 256, 256, 0, stream>>>(WQ, wq16, NW);
  cvt_f32_to_f16_kernel<<<(NW + 255) / 256, 256, 0, stream>>>(WK, wk16, NW);
  cvt_f32_to_f16_kernel<<<(NW + 255) / 256, 256, 0, stream>>>(WV, wv16, NW);
  cvt_f32_to_f16_kernel<<<(NW + 255) / 256, 256, 0, stream>>>(WO, wo16, NW);

  // 2) projections: [8192,1024] x [1024,1024] -> head-major f16
  dim3 gProj(M / 128, DMODEL / 128, 1);
  gemm_wmma_kernel<128,128,64, 2,4, 0,0,1><<<gProj, 256, 0, stream>>>(
      inq16, wq16, Qh, nullptr, DMODEL, DMODEL, DMODEL, 0, 0, 0, 0, 0.0f);
  gemm_wmma_kernel<128,128,64, 2,4, 0,0,1><<<gProj, 256, 0, stream>>>(
      ink16, wk16, Kh, nullptr, DMODEL, DMODEL, DMODEL, 0, 0, 0, 0, 0.0f);
  gemm_wmma_kernel<128,128,64, 2,4, 0,0,1><<<gProj, 256, 0, stream>>>(
      inv16, wv16, Vh, nullptr, DMODEL, DMODEL, DMODEL, 0, 0, 0, 0, 0.0f);

  // 3) scores per (b,h): Q[2048,64] x K^T -> attn region of d_out
  //    quirk: scale = 1/sqrt(S) = 1/sqrt(2048)
  const float scale = 0.022097086912079608f;
  dim3 gScore(SEQ / 128, SEQ / 128, 4 * NHEAD);
  gemm_wmma_kernel<128,128,64, 2,4, 0,1,2><<<gScore, 256, 0, stream>>>(
      Qh, Kh, attn, mask, HDIM, HDIM, HDIM, SEQ,
      (long long)SEQ * HDIM, (long long)SEQ * HDIM, (long long)SEQ * SEQ, scale);

  // 4) softmax in place over last dim (2048), one block per row
  softmax_rows_kernel<<<4 * NHEAD * SEQ, 256, 0, stream>>>(attn);

  // 5) context per (b,h): attn(f32)[2048,2048] x V[2048,64] -> ctx16 [B,S,1024]
  dim3 gCtx(SEQ / 128, 1, 4 * NHEAD);
  gemm_wmma_kernel<128,64,64, 4,2, 1,0,3><<<gCtx, 256, 0, stream>>>(
      attn, Vh, ctx16, nullptr, SEQ, SEQ, HDIM, 0,
      (long long)SEQ * SEQ, (long long)SEQ * HDIM, 0, 0.0f);

  // 6) output projection: ctx16 [8192,1024] x Wo -> proj f32
  gemm_wmma_kernel<128,128,64, 2,4, 0,0,0><<<gProj, 256, 0, stream>>>(
      ctx16, wo16, proj, nullptr, DMODEL, DMODEL, DMODEL, DMODEL, 0, 0, 0, 0.0f);

  // 7) residual + LayerNorm -> d_out[:8M]
  residual_ln_kernel<<<M, 256, 0, stream>>>(proj, inQ, out);
}
